// EncoderLayer_42606075576906
// MI455X (gfx1250) — matrix operand order repaired and checked by reading, not compiled, not run on hardware
//
#include <hip/hip_runtime.h>

// ---------------------------------------------------------------------------
// Transformer encoder layer for MI455X (gfx1250), bf16 WMMA + f32 accumulate,
// with Tensor Data Mover (TDM) async staging of GEMM A-tiles.
// D=1024, H=16, DK=64, DFF=4096, B=2, S=2048  -> M = B*S = 4096 rows.
// ---------------------------------------------------------------------------

typedef __bf16 bf16;
typedef __bf16 bf16x8  __attribute__((ext_vector_type(8)));
typedef __bf16 bf16x16 __attribute__((ext_vector_type(16)));
typedef float  f32x8   __attribute__((ext_vector_type(8)));
typedef unsigned int v4u __attribute__((ext_vector_type(4)));
typedef int          v4i __attribute__((ext_vector_type(4)));
typedef int          v8i __attribute__((ext_vector_type(8)));

#define D_MODEL 1024
#define NHEAD   16
#define HDK     64
#define DFF     4096
#define BATCH   2
#define SEQ     2048
#define ROWS    (BATCH * SEQ)   // 4096
#define EPS     1e-6f
#define NEG_SLOPE 0.01f

// ---------------------------------------------------------------------------
// TDM: async 2D tile load (global -> LDS), bf16 elements, row-major tile.
// Builds the Tensor DMA Descriptor (D#) per CDNA5 ISA:
//   group0: [1:0]=count=1, [63:32]=lds_addr, [120:64]=global_addr, [127:126]=2
//   group1: [17:16]=data_size(1 => 2B), [79:48]=tensor_dim0, [111:80]=tensor_dim1,
//           [127:112]=tile_dim0, [143:128]=tile_dim1, [207:160]=dim0_stride
// Tensor dims set huge so no OOB clipping occurs (tile is always in-bounds).
// Tracked with TENSORcnt; caller pairs with s_wait_tensorcnt + barrier.
// This toolchain's builtin takes 6 args (clang-23 / therock-10.0 form):
//   (uint32x4 g0, int32x8 g1, int32x4 g2, int32x4 g3, int32x8 pad, i32 cpol)
// ---------------------------------------------------------------------------
__device__ inline void tdm_load_tile_2d(const void* gaddr, unsigned lds_off,
                                        unsigned tile_w, unsigned tile_h,
                                        unsigned stride_elems)
{
    unsigned long long ga = (unsigned long long)(size_t)gaddr;
    v4u g0;
    g0[0] = 1u;                                            // count = 1 (valid)
    g0[1] = lds_off;                                       // LDS byte address
    g0[2] = (unsigned)ga;                                  // global_addr[31:0]
    g0[3] = (unsigned)((ga >> 32) & 0x01FFFFFFu) | (2u << 30); // [56:32] | type=2

    const unsigned td0 = 0x40000000u;                      // huge tensor dim 0
    const unsigned td1 = 0x40000000u;                      // huge tensor dim 1
    v8i g1;
    g1[0] = (1 << 16);                                     // data_size=1 (2 bytes)
    g1[1] = (int)((td0 & 0xFFFFu) << 16);                  // tensor_dim0[15:0]
    g1[2] = (int)(((td0 >> 16) & 0xFFFFu) | ((td1 & 0xFFFFu) << 16));
    g1[3] = (int)(((td1 >> 16) & 0xFFFFu) | ((tile_w & 0xFFFFu) << 16)); // tile_dim0
    g1[4] = (int)(tile_h & 0xFFFFu);                       // tile_dim1
    g1[5] = (int)stride_elems;                             // tensor_dim0_stride[31:0]
    g1[6] = 0;                                             // stride[47:32], dim1_stride
    g1[7] = 0;

    v4i z4; z4[0] = z4[1] = z4[2] = z4[3] = 0;             // groups 2/3 unused (2D)
    v8i z8;
    #pragma unroll
    for (int i = 0; i < 8; i++) z8[i] = 0;
    __builtin_amdgcn_tensor_load_to_lds(g0, g1, z4, z4, z8, 0);
}

// ---------------------------------------------------------------------------
// f32 -> bf16 elementwise convert (grid-stride)
// ---------------------------------------------------------------------------
__global__ __launch_bounds__(256)
void cvt_f32_bf16(const float* __restrict__ in, bf16* __restrict__ out, int n)
{
    int i = blockIdx.x * blockDim.x + threadIdx.x;
    int stride = gridDim.x * blockDim.x;
    for (; i < n; i += stride) out[i] = (bf16)in[i];
}

// ---------------------------------------------------------------------------
// GEMM: C[M,N] = A[M,K] @ B[K,N] + bias, A/B bf16 row-major, f32 accum.
// Block tile 128x128, BK=32, 256 threads = 8 waves, wave tile 32x64.
// Double-buffered LDS; A-tiles DMA'd by the TDM (wave 0 issues, TENSORcnt
// waited before the block barrier), B-tiles staged + transposed by all waves
// so B-fragments are two contiguous ds_load_b128 per tile.
// Fragment layouts (wave32, v_wmma_f32_16x16x32_bf16):
//   A frag : lane r=lane&15 holds row r; k {0..15} (lanes<16)
//            or {16..31} (lanes>=16)
//   B frag : lane holds column c=lane&15; k = (lane>=16? 16:0)+0..15
//   C frag : VGPR i holds row i + 8*(lane>=16), col = lane&15
// ---------------------------------------------------------------------------
template<bool LEAKY>
__global__ __launch_bounds__(256)
void gemm_bf16_wmma(const bf16* __restrict__ A, const bf16* __restrict__ B,
                    const float* __restrict__ bias,
                    float* __restrict__ outF, bf16* __restrict__ outB,
                    int M, int N, int K)
{
    __shared__ bf16 sA[2][128 * 32];   // [buf][row][k]   (TDM destination)
    __shared__ bf16 sBT[2][128 * 32];  // [buf][col][k]   (transposed)

    const int tid  = threadIdx.x;
    const int lane = tid & 31;
    const int wid  = tid >> 5;
    const int c    = lane & 15;
    const int hh   = lane >> 4;          // half-wave select
    const int wm   = (wid >> 1) * 32;    // wave row offset within block tile
    const int wn   = (wid & 1) * 64;     // wave col offset within block tile
    const int bm   = blockIdx.y * 128;
    const int bn   = blockIdx.x * 128;

    f32x8 acc[2][4];
    #pragma unroll
    for (int i = 0; i < 2; i++)
        #pragma unroll
        for (int j = 0; j < 4; j++)
            #pragma unroll
            for (int e = 0; e < 8; e++) acc[i][j][e] = 0.0f;

    // stage B tile (32 x 128) transposed into sBT[buf][col][k]
    auto stageB = [&](int k0, int buf) {
        #pragma unroll
        for (int i = 0; i < 2; i++) {
            int lin = tid + i * 256;          // 0..511
            int kr = lin >> 4, seg = lin & 15;
            bf16x8 v = *(const bf16x8*)&B[(size_t)(k0 + kr) * N + bn + seg * 8];
            #pragma unroll
            for (int j = 0; j < 8; j++) sBT[buf][(seg * 8 + j) * 32 + kr] = v[j];
        }
    };

    const int nk = K / 32;

    // --- prologue: prefetch tile 0 ---------------------------------------
    if (wid == 0)
        tdm_load_tile_2d(A + (size_t)bm * K, (unsigned)(size_t)&sA[0][0],
                         /*tile_w=*/32, /*tile_h=*/128, /*stride=*/(unsigned)K);
    stageB(0, 0);
    if (wid == 0) __builtin_amdgcn_s_wait_tensorcnt(0);
    __syncthreads();

    for (int it = 0; it < nk; it++) {
        const int cur = it & 1, nxt = cur ^ 1;

        // --- prefetch tile it+1 into the other buffer (overlaps compute) -
        if (it + 1 < nk) {
            if (wid == 0)
                tdm_load_tile_2d(A + (size_t)bm * K + (size_t)(it + 1) * 32,
                                 (unsigned)(size_t)&sA[nxt][0],
                                 32, 128, (unsigned)K);
            stageB((it + 1) * 32, nxt);
        }

        // --- B fragments (4 N-tiles) -------------------------------------
        bf16x16 bf[4];
        #pragma unroll
        for (int nt = 0; nt < 4; nt++) {
            const bf16* p = &sBT[cur][(wn + nt * 16 + c) * 32 + hh * 16];
            union { uint4 u[2]; bf16x16 v; } t;
            t.u[0] = *(const uint4*)p;
            t.u[1] = *(const uint4*)(p + 8);
            bf[nt] = t.v;
        }
        // --- A fragments + WMMA ------------------------------------------
        #pragma unroll
        for (int mt = 0; mt < 2; mt++) {
            const bf16* pa = &sA[cur][(wm + mt * 16 + c) * 32 + hh * 16];
            union { uint4 u[2]; bf16x16 v; } ta;
            ta.u[0] = *(const uint4*)pa;        // k {0..7} or {16..23}
            ta.u[1] = *(const uint4*)(pa + 8);  // k {8..15} or {24..31}
            bf16x16 af = ta.v;
            #pragma unroll
            for (int nt = 0; nt < 4; nt++)
                acc[mt][nt] = __builtin_amdgcn_wmma_f32_16x16x32_bf16(
                    false, af, false, bf[nt], (short)0, acc[mt][nt], false, false);
        }

        if (wid == 0) __builtin_amdgcn_s_wait_tensorcnt(0);
        __syncthreads();
    }

    // --- epilogue: bias (+leaky), write f32 and/or bf16 ------------------
    #pragma unroll
    for (int mt = 0; mt < 2; mt++) {
        #pragma unroll
        for (int nt = 0; nt < 4; nt++) {
            int col = bn + wn + nt * 16 + c;
            float bv = bias ? bias[col] : 0.0f;
            #pragma unroll
            for (int i = 0; i < 8; i++) {
                int row = bm + wm + mt * 16 + i + 8 * hh;
                float v = acc[mt][nt][i] + bv;
                if (LEAKY) v = v > 0.0f ? v : NEG_SLOPE * v;
                size_t idx = (size_t)row * N + col;
                if (outF) outF[idx] = v;
                if (outB) outB[idx] = (bf16)v;
            }
        }
    }
}

// ---------------------------------------------------------------------------
// Flash attention: one block = one (b,h) x 64 query rows; 4 waves, each wave
// owns 16 query rows.  Loops over 32-key tiles: S = Q K^T (2 N-tiles x 2
// k-steps WMMA), online softmax across the accumulator layout, P bounced
// through per-wave LDS into A-fragment layout, ctx += P V (4 WMMAs against
// LDS-transposed V).  Q/K/V are bf16 [B,S,D] with head offset h*64.
// ---------------------------------------------------------------------------
__global__ __launch_bounds__(128)
void flash_attn_wmma(const bf16* __restrict__ Q, const bf16* __restrict__ Km,
                     const bf16* __restrict__ V, bf16* __restrict__ O)
{
    __shared__ bf16 sVT[64 * 32];       // [d][key] transposed V tile
    __shared__ bf16 sP[4][16 * 32];     // per-wave P tile [row][key]

    const int tid  = threadIdx.x;
    const int lane = tid & 31;
    const int wid  = tid >> 5;
    const int c    = lane & 15;
    const int hh   = lane >> 4;
    const int qt = blockIdx.x, h = blockIdx.y, b = blockIdx.z;
    const int q0 = qt * 64 + wid * 16;
    const size_t hoff = (size_t)h * HDK;
    const size_t bs   = (size_t)b * SEQ;

    // preload Q A-fragments (DK=64 -> two 32-wide k-steps)
    bf16x16 qf[2];
    {
        const bf16* pq = Q + (bs + q0 + c) * D_MODEL + hoff;
        #pragma unroll
        for (int ks = 0; ks < 2; ks++) {
            union { uint4 u[2]; bf16x16 v; } t;
            t.u[0] = *(const uint4*)(pq + ks * 32 + hh * 16);
            t.u[1] = *(const uint4*)(pq + ks * 32 + hh * 16 + 8);
            qf[ks] = t.v;
        }
    }

    float m[8], l[8];
    f32x8 o[4];
    #pragma unroll
    for (int i = 0; i < 8; i++) { m[i] = -1e30f; l[i] = 0.0f; }
    #pragma unroll
    for (int f = 0; f < 4; f++)
        #pragma unroll
        for (int i = 0; i < 8; i++) o[f][i] = 0.0f;

    for (int kt = 0; kt < SEQ / 32; kt++) {
        const int key0 = kt * 32;
        __syncthreads();   // protect sVT from previous iteration's readers
        // stage V tile transposed: sVT[d][key]
        #pragma unroll
        for (int i = 0; i < 2; i++) {
            int lin = tid + i * 128;           // 0..255
            int key = lin >> 3, seg = lin & 7;
            bf16x8 v = *(const bf16x8*)&V[(bs + key0 + key) * D_MODEL + hoff + seg * 8];
            #pragma unroll
            for (int j = 0; j < 8; j++) sVT[(seg * 8 + j) * 32 + key] = v[j];
        }
        __syncthreads();

        // scores: S[16 x 32] = Q_tile (16x64) @ K_tile^T (64x32)
        f32x8 s[2];
        #pragma unroll
        for (int nt = 0; nt < 2; nt++) {
            #pragma unroll
            for (int e = 0; e < 8; e++) s[nt][e] = 0.0f;
            // B frag: column = key row of K -> contiguous 16 bf16 of K[key,:]
            const bf16* pk = Km + (bs + key0 + nt * 16 + c) * D_MODEL + hoff + hh * 16;
            #pragma unroll
            for (int ks = 0; ks < 2; ks++) {
                union { uint4 u[2]; bf16x16 v; } t;
                t.u[0] = *(const uint4*)(pk + ks * 32);
                t.u[1] = *(const uint4*)(pk + ks * 32 + 8);
                s[nt] = __builtin_amdgcn_wmma_f32_16x16x32_bf16(
                    false, qf[ks], false, t.v, (short)0, s[nt], false, false);
            }
        }

        // online softmax; row r = i + 8*hh, reduce over 16 lanes of same hh
        #pragma unroll
        for (int i = 0; i < 8; i++) {
            float s0 = s[0][i] * 0.125f;   // 1/sqrt(64)
            float s1 = s[1][i] * 0.125f;
            float mx = fmaxf(s0, s1);
            #pragma unroll
            for (int off = 1; off < 16; off <<= 1)
                mx = fmaxf(mx, __shfl_xor(mx, off, 32));
            float mn = fmaxf(m[i], mx);
            float sc = __expf(m[i] - mn);
            float p0 = __expf(s0 - mn);
            float p1 = __expf(s1 - mn);
            float rs = p0 + p1;
            #pragma unroll
            for (int off = 1; off < 16; off <<= 1)
                rs += __shfl_xor(rs, off, 32);
            l[i] = l[i] * sc + rs;
            m[i] = mn;
            #pragma unroll
            for (int f = 0; f < 4; f++) o[f][i] *= sc;
            sP[wid][(i + 8 * hh) * 32 + c]      = (bf16)p0;
            sP[wid][(i + 8 * hh) * 32 + 16 + c] = (bf16)p1;
        }
        __syncthreads();

        // ctx += P (16x32) @ V_tile (32x64)
        union { uint4 u[2]; bf16x16 v; } tp;
        const bf16* pp = &sP[wid][c * 32 + hh * 16];
        tp.u[0] = *(const uint4*)pp;
        tp.u[1] = *(const uint4*)(pp + 8);
        bf16x16 pf = tp.v;
        #pragma unroll
        for (int f = 0; f < 4; f++) {
            const bf16* pv = &sVT[(f * 16 + c) * 32 + hh * 16];
            union { uint4 u[2]; bf16x16 v; } tv;
            tv.u[0] = *(const uint4*)pv;
            tv.u[1] = *(const uint4*)(pv + 8);
            o[f] = __builtin_amdgcn_wmma_f32_16x16x32_bf16(
                false, pf, false, tv.v, (short)0, o[f], false, false);
        }
    }

    // normalize and write ctx (bf16 [B,S,D], reassembled per head)
    #pragma unroll
    for (int f = 0; f < 4; f++) {
        int d = f * 16 + c;
        #pragma unroll
        for (int i = 0; i < 8; i++) {
            int q = q0 + i + 8 * hh;
            O[(bs + q) * D_MODEL + hoff + d] = (bf16)(o[f][i] / l[i]);
        }
    }
}

// ---------------------------------------------------------------------------
// out = LayerNorm(a + r) * gamma + beta ; one row per block (D=1024, 256 thr)
// ---------------------------------------------------------------------------
__global__ __launch_bounds__(256)
void residual_layernorm(const float* __restrict__ a, const float* __restrict__ r,
                        const float* __restrict__ gamma, const float* __restrict__ beta,
                        float* __restrict__ outF, bf16* __restrict__ outB)
{
    __shared__ float red[16];
    const int row = blockIdx.x, tid = threadIdx.x;
    const size_t base = (size_t)row * D_MODEL;

    float4 av = ((const float4*)(a + base))[tid];
    float4 rv = ((const float4*)(r + base))[tid];
    float v0 = av.x + rv.x, v1 = av.y + rv.y, v2 = av.z + rv.z, v3 = av.w + rv.w;
    float s = v0 + v1 + v2 + v3;
    float q = v0 * v0 + v1 * v1 + v2 * v2 + v3 * v3;
    #pragma unroll
    for (int off = 1; off < 32; off <<= 1) {
        s += __shfl_xor(s, off, 32);
        q += __shfl_xor(q, off, 32);
    }
    const int wid = tid >> 5, lane = tid & 31;
    if (lane == 0) { red[wid] = s; red[8 + wid] = q; }
    __syncthreads();
    float ts = 0.0f, tq = 0.0f;
    #pragma unroll
    for (int i = 0; i < 8; i++) { ts += red[i]; tq += red[8 + i]; }
    float mean = ts * (1.0f / D_MODEL);
    float var  = tq * (1.0f / D_MODEL) - mean * mean;
    float rstd = rsqrtf(var + EPS);

    int col = tid * 4;
    float o0 = (v0 - mean) * rstd * gamma[col + 0] + beta[col + 0];
    float o1 = (v1 - mean) * rstd * gamma[col + 1] + beta[col + 1];
    float o2 = (v2 - mean) * rstd * gamma[col + 2] + beta[col + 2];
    float o3 = (v3 - mean) * rstd * gamma[col + 3] + beta[col + 3];
    float4 ov = { o0, o1, o2, o3 };
    ((float4*)(outF + base))[tid] = ov;
    if (outB) {
        outB[base + col + 0] = (bf16)o0;
        outB[base + col + 1] = (bf16)o1;
        outB[base + col + 2] = (bf16)o2;
        outB[base + col + 3] = (bf16)o3;
    }
}

// ---------------------------------------------------------------------------
// Orchestration
// ---------------------------------------------------------------------------
extern "C" void kernel_launch(void* const* d_in, const int* in_sizes, int n_in,
                              void* d_out, int out_size, void* d_ws, size_t ws_size,
                              hipStream_t stream)
{
    (void)in_sizes; (void)n_in; (void)out_size; (void)ws_size;

    const float* x     = (const float*)d_in[0];
    const float* wq    = (const float*)d_in[1];
    const float* bq    = (const float*)d_in[2];
    const float* wk    = (const float*)d_in[3];
    const float* bk    = (const float*)d_in[4];
    const float* wv    = (const float*)d_in[5];
    const float* bv    = (const float*)d_in[6];
    const float* wo    = (const float*)d_in[7];
    const float* bo    = (const float*)d_in[8];
    const float* g1v   = (const float*)d_in[9];
    const float* b1v   = (const float*)d_in[10];
    const float* w_ff1 = (const float*)d_in[11];
    const float* b_ff1 = (const float*)d_in[12];
    const float* w_ff2 = (const float*)d_in[13];
    const float* b_ff2 = (const float*)d_in[14];
    const float* g2v   = (const float*)d_in[15];
    const float* b2v   = (const float*)d_in[16];

    // --- workspace carve-up (deterministic) ------------------------------
    char* ws = (char*)d_ws;
    size_t off = 0;
    auto alloc = [&](size_t bytes) -> void* {
        void* p = ws + off;
        off += (bytes + 255) & ~(size_t)255;
        return p;
    };
    const size_t MD  = (size_t)ROWS * D_MODEL;     // 4M elems
    const size_t DD  = (size_t)D_MODEL * D_MODEL;  // 1M elems
    const size_t DF  = (size_t)D_MODEL * DFF;      // 4M elems
    const size_t MF  = (size_t)ROWS * DFF;         // 16M elems

    bf16* xb   = (bf16*)alloc(MD * 2);
    bf16* wqb  = (bf16*)alloc(DD * 2);
    bf16* wkb  = (bf16*)alloc(DD * 2);
    bf16* wvb  = (bf16*)alloc(DD * 2);
    bf16* wob  = (bf16*)alloc(DD * 2);
    bf16* w1b  = (bf16*)alloc(DF * 2);
    bf16* w2b  = (bf16*)alloc(DF * 2);
    bf16* qb   = (bf16*)alloc(MD * 2);
    bf16* kb   = (bf16*)alloc(MD * 2);
    bf16* vb   = (bf16*)alloc(MD * 2);
    bf16* ctxb = (bf16*)alloc(MD * 2);
    float* tmpF = (float*)alloc(MD * 4);           // attn_out, reused for ffn2 out
    float* hF   = (float*)alloc(MD * 4);
    bf16*  hb   = (bf16*)alloc(MD * 2);
    bf16*  ffb  = (bf16*)alloc(MF * 2);

    // --- 1. bf16 down-converts ------------------------------------------
    cvt_f32_bf16<<<2048, 256, 0, stream>>>(x,     xb,  (int)MD);
    cvt_f32_bf16<<<1024, 256, 0, stream>>>(wq,    wqb, (int)DD);
    cvt_f32_bf16<<<1024, 256, 0, stream>>>(wk,    wkb, (int)DD);
    cvt_f32_bf16<<<1024, 256, 0, stream>>>(wv,    wvb, (int)DD);
    cvt_f32_bf16<<<1024, 256, 0, stream>>>(wo,    wob, (int)DD);
    cvt_f32_bf16<<<2048, 256, 0, stream>>>(w_ff1, w1b, (int)DF);
    cvt_f32_bf16<<<2048, 256, 0, stream>>>(w_ff2, w2b, (int)DF);

    dim3 blk(256);
    dim3 gridDxD(D_MODEL / 128, ROWS / 128);   // (8, 32)
    dim3 gridDFF(DFF / 128, ROWS / 128);       // (32, 32)

    // --- 2. Q/K/V projections (bf16 out) --------------------------------
    gemm_bf16_wmma<false><<<gridDxD, blk, 0, stream>>>(xb, wqb, bq, nullptr, qb,
                                                       ROWS, D_MODEL, D_MODEL);
    gemm_bf16_wmma<false><<<gridDxD, blk, 0, stream>>>(xb, wkb, bk, nullptr, kb,
                                                       ROWS, D_MODEL, D_MODEL);
    gemm_bf16_wmma<false><<<gridDxD, blk, 0, stream>>>(xb, wvb, bv, nullptr, vb,
                                                       ROWS, D_MODEL, D_MODEL);

    // --- 3. flash attention ---------------------------------------------
    flash_attn_wmma<<<dim3(SEQ / 64, NHEAD, BATCH), 128, 0, stream>>>(qb, kb, vb, ctxb);

    // --- 4. output projection (f32 out) ---------------------------------
    gemm_bf16_wmma<false><<<gridDxD, blk, 0, stream>>>(ctxb, wob, bo, tmpF, nullptr,
                                                       ROWS, D_MODEL, D_MODEL);

    // --- 5. residual + LN1 ----------------------------------------------
    residual_layernorm<<<ROWS, 256, 0, stream>>>(tmpF, x, g1v, b1v, hF, hb);

    // --- 6. FFN1 + LeakyReLU (bf16 out) ---------------------------------
    gemm_bf16_wmma<true><<<gridDFF, blk, 0, stream>>>(hb, w1b, b_ff1, nullptr, ffb,
                                                      ROWS, DFF, D_MODEL);

    // --- 7. FFN2 (f32 out) ----------------------------------------------
    gemm_bf16_wmma<false><<<gridDxD, blk, 0, stream>>>(ffb, w2b, b_ff2, tmpF, nullptr,
                                                       ROWS, D_MODEL, DFF);

    // --- 8. residual + LN2 -> d_out -------------------------------------
    residual_layernorm<<<ROWS, 256, 0, stream>>>(tmpF, hF, g2v, b2v,
                                                 (float*)d_out, nullptr);
}
